// CrossAttention_82686710383187
// MI455X (gfx1250) — matrix operand order repaired and checked by reading, not compiled, not run on hardware
//
#include <hip/hip_runtime.h>

// ---------------------------------------------------------------------------
// Problem constants (from reference): B=2, N=M=2048, DIM=INNER=512, H=8, DH=64
// ---------------------------------------------------------------------------
#define BQ    2
#define NQ    2048
#define MK    2048
#define DIMC  512
#define HN    8
#define DHD   64
#define INNER 512
#define LN_EPS 1e-5f
// SCALE applied to q and to k in reference -> fold SCALE*SCALE = 1/64 into Q.
#define QK_SCALE (1.0f / 64.0f)

typedef __bf16    bf16x16 __attribute__((ext_vector_type(16)));
typedef float     f32x8   __attribute__((ext_vector_type(8)));
typedef unsigned int u32x4 __attribute__((ext_vector_type(4)));

union FragB {            // 16 bf16 = 32 bytes = 8 VGPRs (A or B operand)
    u32x4   u[2];
    bf16x16 v;
};

__device__ __forceinline__ f32x8 wmma_bf16(bf16x16 a, bf16x16 b, f32x8 c) {
    return __builtin_amdgcn_wmma_f32_16x16x32_bf16(
        /*neg_a=*/false, a, /*neg_b=*/false, b,
        /*c_mod=*/(short)0, c, /*reuse_a=*/false, /*reuse_b=*/false);
}

__device__ __forceinline__ f32x8 zero8() {
    f32x8 z = {0.f, 0.f, 0.f, 0.f, 0.f, 0.f, 0.f, 0.f};
    return z;
}

// ---------------------------------------------------------------------------
// LayerNorm (512 cols), one block per row, 256 threads, bf16 output
// ---------------------------------------------------------------------------
__global__ __launch_bounds__(256) void ln_bf16_kernel(
    const float* __restrict__ x, const float* __restrict__ g,
    const float* __restrict__ bta, __bf16* __restrict__ out)
{
    __shared__ float s1[256], s2[256];
    const int row = blockIdx.x, t = threadIdx.x;
    const float* xr = x + (long)row * DIMC;
    float a = xr[t], b = xr[t + 256];
    s1[t] = a + b;
    s2[t] = a * a + b * b;
    __syncthreads();
    for (int off = 128; off; off >>= 1) {
        if (t < off) { s1[t] += s1[t + off]; s2[t] += s2[t + off]; }
        __syncthreads();
    }
    const float mu  = s1[0] * (1.0f / DIMC);
    const float var = s2[0] * (1.0f / DIMC) - mu * mu;
    const float rs  = rsqrtf(var + LN_EPS);
    out[(long)row * DIMC + t]       = (__bf16)((a - mu) * rs * g[t] + bta[t]);
    out[(long)row * DIMC + t + 256] = (__bf16)((b - mu) * rs * g[t + 256] + bta[t + 256]);
}

// Same but f32 output (final LN writes d_out)
__global__ __launch_bounds__(256) void ln_f32_kernel(
    const float* __restrict__ x, const float* __restrict__ g,
    const float* __restrict__ bta, float* __restrict__ out)
{
    __shared__ float s1[256], s2[256];
    const int row = blockIdx.x, t = threadIdx.x;
    const float* xr = x + (long)row * DIMC;
    float a = xr[t], b = xr[t + 256];
    s1[t] = a + b;
    s2[t] = a * a + b * b;
    __syncthreads();
    for (int off = 128; off; off >>= 1) {
        if (t < off) { s1[t] += s1[t + off]; s2[t] += s2[t + off]; }
        __syncthreads();
    }
    const float mu  = s1[0] * (1.0f / DIMC);
    const float var = s2[0] * (1.0f / DIMC) - mu * mu;
    const float rs  = rsqrtf(var + LN_EPS);
    out[(long)row * DIMC + t]       = (a - mu) * rs * g[t] + bta[t];
    out[(long)row * DIMC + t + 256] = (b - mu) * rs * g[t + 256] + bta[t + 256];
}

// ---------------------------------------------------------------------------
// Elementwise f32 -> bf16 (optional scale)
// ---------------------------------------------------------------------------
__global__ __launch_bounds__(256) void cvt_bf16_kernel(
    const float* __restrict__ in, __bf16* __restrict__ out, int n, float scale)
{
    const int i = blockIdx.x * 256 + threadIdx.x;
    if (i < n) out[i] = (__bf16)(in[i] * scale);
}

// Weight transpose + convert: W[K,N] f32 -> Wt[N,K] bf16
__global__ __launch_bounds__(256) void wtrans_kernel(
    const float* __restrict__ W, __bf16* __restrict__ Wt, int K, int Nc)
{
    const int idx = blockIdx.x * 256 + threadIdx.x;
    if (idx >= K * Nc) return;
    const int k = idx % K, n = idx / K;
    Wt[(long)n * K + k] = (__bf16)W[(long)k * Nc + n];
}

// Extract K half of KV gemm output -> bf16 [B*M, 512]
__global__ __launch_bounds__(256) void extract_k_kernel(
    const float* __restrict__ KV, __bf16* __restrict__ Kb, int n)
{
    const int i = blockIdx.x * 256 + threadIdx.x;
    if (i >= n) return;
    const int row = i >> 9, col = i & 511;
    Kb[i] = (__bf16)KV[(long)row * 1024 + col];
}

// V half of KV gemm output -> transposed bf16 Vt[B*H][64][M]
__global__ __launch_bounds__(256) void make_vt_kernel(
    const float* __restrict__ KV, __bf16* __restrict__ Vt, int n)
{
    const int i = blockIdx.x * 256 + threadIdx.x;  // i = (bh*64 + d)*M + m
    if (i >= n) return;
    const int m  = i & (MK - 1);
    const int d  = (i >> 11) & 63;
    const int bh = i >> 17;
    const int b = bh >> 3, h = bh & 7;
    Vt[i] = (__bf16)KV[((long)b * MK + m) * 1024 + 512 + h * DHD + d];
}

// ---------------------------------------------------------------------------
// Generic WMMA bf16 GEMM: C[Mr,Nc] f32 = A[Mr,K] bf16 @ Wt[Nc,K] bf16 (B^T form)
// Block = 256 threads (8 waves). Wave w: rows 16w..16w+15, 64 cols.
// Grid: (Mr/128, Nc/64)
// ---------------------------------------------------------------------------
__global__ __launch_bounds__(256) void gemm_bf16_kernel(
    const __bf16* __restrict__ A, const __bf16* __restrict__ Wt,
    float* __restrict__ C, int Mr, int Nc, int K)
{
    const int wave = threadIdx.x >> 5, lane = threadIdx.x & 31;
    const int grp = lane >> 4, ln16 = lane & 15;
    const int row0 = blockIdx.x * 128 + wave * 16;
    const int col0 = blockIdx.y * 64;

    f32x8 acc[4];
#pragma unroll
    for (int c = 0; c < 4; ++c) acc[c] = zero8();

    for (int k0 = 0; k0 < K; k0 += 32) {
        FragB a;   // A fragment: row = row0+ln16; halves [grp*8..+7] and [16+grp*8..+7]
        {
            const __bf16* ap = A + (long)(row0 + ln16) * K + k0 + grp * 8;
            a.u[0] = *(const u32x4*)ap;
            a.u[1] = *(const u32x4*)(ap + 16);
        }
#pragma unroll
        for (int c = 0; c < 4; ++c) {
            FragB bf;  // B fragment: col = col0+c*16+ln16; halves [grp*8..+7] and [16+grp*8..+7]
            const __bf16* bp = Wt + (long)(col0 + c * 16 + ln16) * K + k0 + grp * 8;
            bf.u[0] = *(const u32x4*)bp;
            bf.u[1] = *(const u32x4*)(bp + 16);
            acc[c] = wmma_bf16(a.v, bf.v, acc[c]);
        }
    }
#pragma unroll
    for (int c = 0; c < 4; ++c)
#pragma unroll
        for (int r = 0; r < 8; ++r)
            C[(long)(row0 + grp * 8 + r) * Nc + col0 + c * 16 + ln16] = acc[c][r];
}

// ---------------------------------------------------------------------------
// Flash attention, bf16 WMMA. Grid: (B*H, N/128), block 256 (8 waves).
// Each wave: 16 query rows; loop keys in chunks of 32.
// Qb pre-scaled by 1/64; Vt is [B*H][64][M]; Ob bf16 [B*N, INNER].
// ---------------------------------------------------------------------------
__global__ __launch_bounds__(256) void attn_kernel(
    const __bf16* __restrict__ Qb, const __bf16* __restrict__ Kb,
    const __bf16* __restrict__ Vt, const unsigned char* __restrict__ msk,
    __bf16* __restrict__ Ob)
{
    const int bh = blockIdx.x;
    const int b = bh >> 3, h = bh & 7;
    const int wave = threadIdx.x >> 5, lane = threadIdx.x & 31;
    const int grp = lane >> 4, ln16 = lane & 15;
    const int qRow0 = blockIdx.y * 128 + wave * 16;

    __shared__ __align__(16) __bf16 pbuf[8][16 * 32];  // per-wave P staging
    __bf16* pw = &pbuf[wave][0];

    // Q A-fragments for d = [0,32) and [32,64)
    FragB qa[2];
    {
        const __bf16* qp = Qb + (long)(b * NQ + qRow0 + ln16) * INNER + h * DHD;
#pragma unroll
        for (int dd = 0; dd < 2; ++dd) {
            const __bf16* p = qp + dd * 32 + grp * 8;
            qa[dd].u[0] = *(const u32x4*)p;
            qa[dd].u[1] = *(const u32x4*)(p + 16);
        }
    }

    f32x8 acc[4];
#pragma unroll
    for (int c = 0; c < 4; ++c) acc[c] = zero8();

    float mrow[8], lrow[8];
#pragma unroll
    for (int r = 0; r < 8; ++r) { mrow[r] = -1e30f; lrow[r] = 0.f; }

    const unsigned char* mp = msk + (long)b * MK;

    for (int j0 = 0; j0 < MK; j0 += 32) {
        // ---- S = Q K^T for two 16-key tiles ----
        f32x8 s[2];
#pragma unroll
        for (int t = 0; t < 2; ++t) {
            f32x8 c = zero8();
#pragma unroll
            for (int dd = 0; dd < 2; ++dd) {
                FragB kb;  // B frag: col = key j0+t*16+ln16; d = dd*32+grp*8..+7 and dd*32+16+grp*8..+7
                const __bf16* kp = Kb + (long)(b * MK + j0 + t * 16 + ln16) * INNER
                                      + h * DHD + dd * 32 + grp * 8;
                kb.u[0] = *(const u32x4*)kp;
                kb.u[1] = *(const u32x4*)(kp + 16);
                c = wmma_bf16(qa[dd].v, kb.v, c);
            }
            // mask: column key index = j0 + t*16 + ln16 (same for all 8 rows)
            if (mp[j0 + t * 16 + ln16] == 0) {
#pragma unroll
                for (int r = 0; r < 8; ++r) c[r] = -1e30f;
            }
            s[t] = c;
        }

        // ---- online softmax row statistics (rows live in VGPR index,
        //      columns across the 16 lanes of each half-wave group) ----
#pragma unroll
        for (int r = 0; r < 8; ++r) {
            float rm = fmaxf(s[0][r], s[1][r]);
#pragma unroll
            for (int off = 8; off; off >>= 1)
                rm = fmaxf(rm, __shfl_xor(rm, off, 16));
            const float mnew  = fmaxf(mrow[r], rm);
            const float alpha = __expf(mrow[r] - mnew);
            mrow[r] = mnew;
            const float p0 = __expf(s[0][r] - mnew);
            const float p1 = __expf(s[1][r] - mnew);
            s[0][r] = p0; s[1][r] = p1;
            float rs = p0 + p1;
#pragma unroll
            for (int off = 8; off; off >>= 1)
                rs += __shfl_xor(rs, off, 16);
            lrow[r] = lrow[r] * alpha + rs;
#pragma unroll
            for (int c = 0; c < 4; ++c) acc[c][r] *= alpha;
        }

        // ---- C-layout P (f32) -> LDS -> A-layout P (bf16) ----
#pragma unroll
        for (int r = 0; r < 8; ++r) {
            pw[(grp * 8 + r) * 32 + ln16]      = (__bf16)s[0][r];
            pw[(grp * 8 + r) * 32 + 16 + ln16] = (__bf16)s[1][r];
        }
        asm volatile("s_wait_dscnt 0" ::: "memory");
        FragB pa;
        {
            const __bf16* pr = pw + ln16 * 32 + grp * 8;
            pa.u[0] = *(const u32x4*)pr;
            pa.u[1] = *(const u32x4*)(pr + 16);
        }

        // ---- O += P @ V  (V B-frags contiguous along keys in Vt) ----
#pragma unroll
        for (int c = 0; c < 4; ++c) {
            FragB vb;  // col = out-dim d = c*16+ln16; keys j0+grp*8..+7 and j0+16+grp*8..+7
            const __bf16* vp = Vt + ((long)bh * DHD + c * 16 + ln16) * MK + j0 + grp * 8;
            vb.u[0] = *(const u32x4*)vp;
            vb.u[1] = *(const u32x4*)(vp + 16);
            acc[c] = wmma_bf16(pa.v, vb.v, acc[c]);
        }
    }

    // ---- epilogue: normalize and emit bf16 rows for the output projection ----
    float invl[8];
#pragma unroll
    for (int r = 0; r < 8; ++r) invl[r] = 1.0f / lrow[r];
#pragma unroll
    for (int c = 0; c < 4; ++c)
#pragma unroll
        for (int r = 0; r < 8; ++r) {
            const long row = (long)b * NQ + qRow0 + grp * 8 + r;
            Ob[row * INNER + h * DHD + c * 16 + ln16] = (__bf16)(acc[c][r] * invl[r]);
        }
}

// ---------------------------------------------------------------------------
// Host orchestration
// ---------------------------------------------------------------------------
extern "C" void kernel_launch(void* const* d_in, const int* in_sizes, int n_in,
                              void* d_out, int out_size, void* d_ws, size_t ws_size,
                              hipStream_t stream)
{
    const float*         x      = (const float*)d_in[0];
    const float*         contex = (const float*)d_in[1];
    const unsigned char* mask   = (const unsigned char*)d_in[2];
    const float*         ln_g   = (const float*)d_in[3];
    const float*         ln_b   = (const float*)d_in[4];
    const float*         Wq     = (const float*)d_in[5];
    const float*         Wkv    = (const float*)d_in[6];
    const float*         Wo     = (const float*)d_in[7];
    const float*         lno_g  = (const float*)d_in[8];
    const float*         lno_b  = (const float*)d_in[9];
    float*               out    = (float*)d_out;

    char* ws = (char*)d_ws;
    size_t off = 0;
    auto carve = [&](size_t bytes) -> char* {
        char* p = ws + off;
        off += (bytes + 255) & ~(size_t)255;
        return p;
    };

    const int ROWS = BQ * NQ;                 // 4096
    __bf16* xnb  = (__bf16*)carve((size_t)ROWS * DIMC * 2);        // LN(x) bf16
    __bf16* ctxb = (__bf16*)carve((size_t)BQ * MK * DIMC * 2);     // context bf16
    __bf16* Wqt  = (__bf16*)carve((size_t)INNER * DIMC * 2);       // [512,512]
    __bf16* Wkvt = (__bf16*)carve((size_t)2 * INNER * DIMC * 2);   // [1024,512]
    __bf16* Wot  = (__bf16*)carve((size_t)DIMC * INNER * 2);       // [512,512]
    float*  Qg   = (float*)carve((size_t)ROWS * INNER * 4);        // q gemm f32
    float*  KVg  = (float*)carve((size_t)BQ * MK * 2 * INNER * 4); // kv gemm f32
    __bf16* Qb   = (__bf16*)carve((size_t)ROWS * INNER * 2);       // q bf16 (scaled)
    __bf16* Kb   = (__bf16*)carve((size_t)BQ * MK * INNER * 2);    // k bf16
    __bf16* Vt   = (__bf16*)carve((size_t)BQ * HN * DHD * MK * 2); // v transposed bf16
    __bf16* Ob   = (__bf16*)carve((size_t)ROWS * INNER * 2);       // attn out bf16
    float*  Og   = (float*)carve((size_t)ROWS * DIMC * 4);         // out proj f32

    // 1. LayerNorm(x) -> bf16
    ln_bf16_kernel<<<ROWS, 256, 0, stream>>>(x, ln_g, ln_b, xnb);

    // 2. context -> bf16
    {
        int n = BQ * MK * DIMC;
        cvt_bf16_kernel<<<(n + 255) / 256, 256, 0, stream>>>(contex, ctxb, n, 1.0f);
    }

    // 3. transpose weights to [N,K] bf16
    wtrans_kernel<<<(DIMC * INNER + 255) / 256, 256, 0, stream>>>(Wq, Wqt, DIMC, INNER);
    wtrans_kernel<<<(DIMC * 2 * INNER + 255) / 256, 256, 0, stream>>>(Wkv, Wkvt, DIMC, 2 * INNER);
    wtrans_kernel<<<(INNER * DIMC + 255) / 256, 256, 0, stream>>>(Wo, Wot, INNER, DIMC);

    // 4. Q = xn @ Wq        [4096 x 512] = [4096 x 512] @ [512 x 512]
    gemm_bf16_kernel<<<dim3(ROWS / 128, INNER / 64), 256, 0, stream>>>(
        xnb, Wqt, Qg, ROWS, INNER, DIMC);

    // 5. KV = ctx @ Wkv     [4096 x 1024]
    gemm_bf16_kernel<<<dim3(ROWS / 128, (2 * INNER) / 64), 256, 0, stream>>>(
        ctxb, Wkvt, KVg, ROWS, 2 * INNER, DIMC);

    // 6. Q -> bf16 with folded softmax scale 1/64
    {
        int n = ROWS * INNER;
        cvt_bf16_kernel<<<(n + 255) / 256, 256, 0, stream>>>(Qg, Qb, n, QK_SCALE);
    }

    // 7. K half -> bf16
    {
        int n = BQ * MK * INNER;
        extract_k_kernel<<<(n + 255) / 256, 256, 0, stream>>>(KVg, Kb, n);
    }

    // 8. V half -> transposed bf16 [B*H][64][M]
    {
        int n = BQ * HN * DHD * MK;
        make_vt_kernel<<<(n + 255) / 256, 256, 0, stream>>>(KVg, Vt, n);
    }

    // 9. flash attention
    attn_kernel<<<dim3(BQ * HN, NQ / 128), 256, 0, stream>>>(Qb, Kb, Vt, mask, Ob);

    // 10. out proj: Og = Ob @ Wo
    gemm_bf16_kernel<<<dim3(ROWS / 128, DIMC / 64), 256, 0, stream>>>(
        Ob, Wot, Og, ROWS, DIMC, INNER);

    // 11. final LayerNorm -> d_out (f32)
    ln_f32_kernel<<<ROWS, 256, 0, stream>>>(Og, lno_g, lno_b, out);

    (void)in_sizes; (void)n_in; (void)out_size; (void)ws_size;
}
